// Dilated_res_block_pooling_angle_35141422416468
// MI455X (gfx1250) — matrix operand order repaired and checked by reading, not compiled, not run on hardware
//
#include <hip/hip_runtime.h>
#include <hip/hip_bf16.h>

// ---------------------------------------------------------------------------
// Dilated_res_block_pooling_angle for MI455X (gfx1250, wave32, WMMA bf16)
//
// B=2, N=40960 -> M = 81920 points, K=16 neighbors, DIN=128, DOUT=128.
// BN folded into bf16 weights + f32 bias. Weights additionally stored
// TRANSPOSED (col-major) so WMMA B fragments are two contiguous 16B loads.
// Neighbor gathers widened to b64 (4 bf16 channels per load).
// ---------------------------------------------------------------------------

typedef __attribute__((ext_vector_type(16))) __bf16 v16bf;
typedef __attribute__((ext_vector_type(8)))  float  v8f;

#define MPTS   81920          // B*N
#define NPTS   40960          // N (per batch)
#define KNB    16             // neighbors

static __device__ __forceinline__ unsigned short f2bf(float f) {
    union { float f; unsigned u; } x; x.f = f;
    unsigned u = x.u;
    unsigned r = u + 0x7fffu + ((u >> 16) & 1u);   // round-to-nearest-even
    return (unsigned short)(r >> 16);
}
static __device__ __forceinline__ float bf2f(unsigned short s) {
    union { unsigned u; float f; } x; x.u = ((unsigned)s) << 16;
    return x.f;
}
// swap lanes r <-> r+16 in a wave32 (ds_swizzle SWAPX16: xor=0x10,and=0x1f)
static __device__ __forceinline__ float swap_half(float x) {
    return __int_as_float(__builtin_amdgcn_ds_swizzle(__float_as_int(x), 0x401f));
}

struct Frag16 {
    union { uint4 q[2]; unsigned short s[16]; v16bf v; };
};

static __device__ __forceinline__ v8f wmma_bf16(const Frag16& a, const Frag16& b, v8f c) {
    return __builtin_amdgcn_wmma_f32_16x16x32_bf16(
        /*neg_a=*/false, a.v, /*neg_b=*/false, b.v,
        /*c_mod=*/(short)0, c, /*reuse_a=*/false, /*reuse_b=*/false);
}

// ---------------------------------------------------------------------------
// BN folding. Writes row-major bf16 W, transposed bf16 Wt[cout][cin], bias.
// ---------------------------------------------------------------------------
__global__ void fold_bn_kernel(const float* __restrict__ w,
                               const float* __restrict__ gamma,
                               const float* __restrict__ beta,
                               const float* __restrict__ mean,
                               const float* __restrict__ var,
                               unsigned short* __restrict__ wbf,
                               unsigned short* __restrict__ wbf_t,
                               float* __restrict__ bias,
                               int total, int cin, int cout) {
    int i = blockIdx.x * 256 + threadIdx.x;
    if (i >= total) return;
    int rw = i / cout, c = i - rw * cout;
    float s = gamma[c] * rsqrtf(var[c] + 1e-5f);
    unsigned short v = f2bf(w[i] * s);
    wbf[i] = v;
    wbf_t[c * cin + rw] = v;
    if (i < cout) bias[i] = beta[i] - mean[i] * s;
}

// feature f32 -> bf16, 2 elements per thread (packed u32 store)
__global__ void cvt_bf16_kernel(const float* __restrict__ in,
                                unsigned* __restrict__ out, int npairs) {
    int i = blockIdx.x * 256 + threadIdx.x;
    if (i < npairs) {
        float a = in[2 * i], b = in[2 * i + 1];
        out[i] = (unsigned)f2bf(a) | ((unsigned)f2bf(b) << 16);
    }
}

// ---------------------------------------------------------------------------
// Generic WMMA GEMM:  C[M x NF] = act(A[M x 128] @ W[128 x NF] + bias)
// 256 threads = 8 waves; each wave owns 16 rows x all NF cols.
// Wt (transposed, col-major) staged in LDS, padded row stride 136 ushorts
// -> B fragment = two ds_load_b128, bank-conflict free.
// ---------------------------------------------------------------------------
#define WPAD 136

template <int NF, bool RELU>
__global__ __launch_bounds__(256) void gemm_bf16_kernel(
        const unsigned short* __restrict__ A,
        const unsigned short* __restrict__ Wt,   // [NF][128]
        const float* __restrict__ bias,
        unsigned short* __restrict__ C) {
    constexpr int K = 128;
    __shared__ __align__(16) unsigned short sWt[64 * WPAD];

    const int tid  = threadIdx.x;
    const int wave = tid >> 5, lane = tid & 31;
    const int half = lane >> 4, r = lane & 15;
    const long row0 = ((long)blockIdx.x * 8 + wave) * 16;

    // A fragments: contiguous k order, same as B (lane<16: K 0-15, lane>=16: K 16-31)
    Frag16 aF[K / 32];
    const uint4* pA = reinterpret_cast<const uint4*>(A + (size_t)(row0 + r) * K);
#pragma unroll
    for (int kt = 0; kt < K / 32; ++kt) {
        aF[kt].q[0] = pA[kt * 4 + half * 2];
        aF[kt].q[1] = pA[kt * 4 + half * 2 + 1];
    }

    for (int ng = 0; ng < NF / 64; ++ng) {
        __syncthreads();
        // stage 64 transposed columns (each K contiguous) with padded stride
        for (int ch = tid; ch < 64 * (K / 8); ch += 256) {
            int c = ch >> 4, cq = ch & 15;
            *reinterpret_cast<uint4*>(sWt + c * WPAD + cq * 8) =
                reinterpret_cast<const uint4*>(Wt + (size_t)(ng * 64 + c) * K)[cq];
        }
        __syncthreads();
#pragma unroll
        for (int t = 0; t < 4; ++t) {
            const int n0 = t * 16;
            v8f acc = {0.f, 0.f, 0.f, 0.f, 0.f, 0.f, 0.f, 0.f};
#pragma unroll
            for (int kt = 0; kt < K / 32; ++kt) {
                Frag16 wF;   // element j = K half*16+j, contiguous in col-major
                const unsigned short* base = sWt + (n0 + r) * WPAD + kt * 32 + half * 16;
                wF.q[0] = *reinterpret_cast<const uint4*>(base);
                wF.q[1] = *reinterpret_cast<const uint4*>(base + 8);
                acc = wmma_bf16(aF[kt], wF, acc);
            }
            const int col = ng * 64 + n0 + r;
            const float bv = bias[col];
#pragma unroll
            for (int i = 0; i < 8; ++i) {
                float v = acc[i] + bv;
                if (RELU) v = fmaxf(v, 0.f);
                C[(size_t)(row0 + half * 8 + i) * NF + col] = f2bf(v);
            }
        }
    }
}

// ---------------------------------------------------------------------------
// LFA stage 1: rel-pos encoding (13) -> lfa_mlp1 (13->64, VALU) -> relu,
// max over 16 neighbors; plus b64 gather-max of f_pc. Writes agg1[M x 128]:
// cols 0-63 = max_k f_pc[idx], cols 64-127 = max_k f_xyz1.
// 256 threads = 4 points x 64 threads.
// ---------------------------------------------------------------------------
__global__ __launch_bounds__(256) void lfa1_kernel(
        const float* __restrict__ xyz, const int* __restrict__ nidx,
        const unsigned short* __restrict__ W1, const float* __restrict__ b1,
        const unsigned short* __restrict__ f_pc,
        unsigned short* __restrict__ agg1) {
    __shared__ unsigned short sW1[13 * 64];
    __shared__ float feat[4][KNB][13];
    __shared__ int   sidx[4][KNB];
    __shared__ float pg[4][4][16][4];   // [pt][kgroup][chquad][4ch]

    const int tid = threadIdx.x;
    const int pt = tid >> 6, c = tid & 63;
    for (int i = tid; i < 13 * 64; i += 256) sW1[i] = W1[i];

    const long p = (long)blockIdx.x * 4 + pt;
    const int  b = (int)(p / NPTS);

    if (c < KNB) {
        const int k = c;
        int iv = nidx[p * KNB + k];
        sidx[pt][k] = iv;
        float px = xyz[p * 3 + 0], py = xyz[p * 3 + 1], pz = xyz[p * 3 + 2];
        long q = (long)b * NPTS + iv;
        float nx = xyz[q * 3 + 0], ny = xyz[q * 3 + 1], nz = xyz[q * 3 + 2];
        float rx = px - nx, ry = py - ny, rz = pz - nz;
        float d  = sqrtf(rx * rx + ry * ry + rz * rz + 1e-12f);
        float iv2 = 1.f / (d + 1e-8f);
        float* f = feat[pt][k];
        f[0] = d;  f[1] = rx; f[2] = ry; f[3] = rz;
        f[4] = px; f[5] = py; f[6] = pz;
        f[7] = nx; f[8] = ny; f[9] = nz;
        f[10] = rx * iv2; f[11] = ry * iv2; f[12] = rz * iv2;
    }
    __syncthreads();

    // f_xyz1 max over neighbors (channel c), coalesced b16 store
    const float bv = b1[c];
    float m2 = -1e30f;
    for (int k = 0; k < KNB; ++k) {
        float acc = bv;
#pragma unroll
        for (int j = 0; j < 13; ++j) acc += feat[pt][k][j] * bf2f(sW1[j * 64 + c]);
        m2 = fmaxf(m2, fmaxf(acc, 0.f));
    }
    agg1[p * 128 + 64 + c] = f2bf(m2);

    // gather-max of f_pc: thread = (channel-quad cq, neighbor-group kg of 4)
    const int cq = c & 15, kg = c >> 4;
    float m[4] = {-1e30f, -1e30f, -1e30f, -1e30f};
    for (int k = kg * 4; k < kg * 4 + 4; ++k) {
        long q = (long)b * NPTS + sidx[pt][k];
        unsigned long long v =
            *reinterpret_cast<const unsigned long long*>(f_pc + q * 64 + cq * 4);
#pragma unroll
        for (int j = 0; j < 4; ++j)
            m[j] = fmaxf(m[j], bf2f((unsigned short)(v >> (16 * j))));
    }
#pragma unroll
    for (int j = 0; j < 4; ++j) pg[pt][kg][cq][j] = m[j];
    __syncthreads();
    if (kg == 0) {
#pragma unroll
        for (int g = 1; g < 4; ++g)
#pragma unroll
            for (int j = 0; j < 4; ++j) m[j] = fmaxf(m[j], pg[pt][g][cq][j]);
        unsigned long long o = 0;
#pragma unroll
        for (int j = 0; j < 4; ++j) o |= (unsigned long long)f2bf(m[j]) << (16 * j);
        *reinterpret_cast<unsigned long long*>(agg1 + p * 128 + cq * 4) = o;
    }
}

// ---------------------------------------------------------------------------
// LFA stage 2 (fused, WMMA): one wave per point.
// Recompute f_xyz1 (16x64, bf16 in LDS), then f_xyz2 = relu(f_xyz1 @ W2 + b2)
// via 8 x v_wmma_f32_16x16x32_bf16; max over rows (neighbors) taken inside the
// accumulator (per-lane + ds_swizzle SWAPX16). W2 fragments preloaded from
// transposed global weights. b64 gather-max of f_aggp fills cols 0-63.
// ---------------------------------------------------------------------------
__global__ __launch_bounds__(256) void lfa2_kernel(
        const float* __restrict__ xyz, const int* __restrict__ nidx,
        const unsigned short* __restrict__ W1, const float* __restrict__ b1,
        const unsigned short* __restrict__ W2t,  // [64][64] transposed
        const float* __restrict__ b2,
        const unsigned short* __restrict__ f_aggp,
        unsigned short* __restrict__ agg2) {
    __shared__ unsigned short sW1[13 * 64];
    __shared__ __align__(16) unsigned short fx1[8][KNB * 64];
    __shared__ float feat[8][KNB][13];
    __shared__ int   sidx[8][KNB];

    const int tid  = threadIdx.x;
    const int wave = tid >> 5, lane = tid & 31;
    const int half = lane >> 4, r = lane & 15;

    for (int i = tid; i < 13 * 64; i += 256) sW1[i] = W1[i];

    // preload all 8 B fragments of W2 (cols n0+r, K contiguous in W2t)
    Frag16 wF2[4][2];
    {
        const uint4* pW = reinterpret_cast<const uint4*>(W2t);
#pragma unroll
        for (int t = 0; t < 4; ++t)
#pragma unroll
            for (int kt = 0; kt < 2; ++kt) {
                int chunk = (t * 16 + r) * 8 + kt * 4 + half * 2;
                wF2[t][kt].q[0] = pW[chunk];
                wF2[t][kt].q[1] = pW[chunk + 1];
            }
    }

    const long p = (long)blockIdx.x * 8 + wave;
    const int  b = (int)(p / NPTS);

    if (lane < KNB) {
        const int k = lane;
        int iv = nidx[p * KNB + k];
        sidx[wave][k] = iv;
        float px = xyz[p * 3 + 0], py = xyz[p * 3 + 1], pz = xyz[p * 3 + 2];
        long q = (long)b * NPTS + iv;
        float nx = xyz[q * 3 + 0], ny = xyz[q * 3 + 1], nz = xyz[q * 3 + 2];
        float rx = px - nx, ry = py - ny, rz = pz - nz;
        float d  = sqrtf(rx * rx + ry * ry + rz * rz + 1e-12f);
        float iv2 = 1.f / (d + 1e-8f);
        float* f = feat[wave][k];
        f[0] = d;  f[1] = rx; f[2] = ry; f[3] = rz;
        f[4] = px; f[5] = py; f[6] = pz;
        f[7] = nx; f[8] = ny; f[9] = nz;
        f[10] = rx * iv2; f[11] = ry * iv2; f[12] = rz * iv2;
    }
    __syncthreads();

    // recompute f_xyz1 tile (16 x 64) in bf16 LDS
#pragma unroll
    for (int t = 0; t < 2; ++t) {
        const int c = lane + t * 32;
        const float bv = b1[c];
        for (int k = 0; k < KNB; ++k) {
            float acc = bv;
#pragma unroll
            for (int j = 0; j < 13; ++j) acc += feat[wave][k][j] * bf2f(sW1[j * 64 + c]);
            fx1[wave][k * 64 + c] = f2bf(fmaxf(acc, 0.f));
        }
    }
    __syncthreads();

    // A fragments (rows = neighbors k, K-dim = 64 -> 2 fragments)
    Frag16 aF[2];
    const uint4* pF = reinterpret_cast<const uint4*>(&fx1[wave][0]);
#pragma unroll
    for (int kt = 0; kt < 2; ++kt) {
        aF[kt].q[0] = pF[r * 8 + kt * 4 + half * 2];
        aF[kt].q[1] = pF[r * 8 + kt * 4 + half * 2 + 1];
    }

#pragma unroll
    for (int t = 0; t < 4; ++t) {
        const int n0 = t * 16;
        v8f acc = {0.f, 0.f, 0.f, 0.f, 0.f, 0.f, 0.f, 0.f};
        acc = wmma_bf16(aF[0], wF2[t][0], acc);
        acc = wmma_bf16(aF[1], wF2[t][1], acc);
        const float bv = b2[n0 + r];
        float m = -1e30f;
#pragma unroll
        for (int i = 0; i < 8; ++i) m = fmaxf(m, fmaxf(acc[i] + bv, 0.f));
        m = fmaxf(m, swap_half(m));            // combine rows 0-7 with 8-15
        if (lane < 16) agg2[p * 128 + 64 + n0 + r] = f2bf(m);
    }

    // gather-max of pooled features (cols 0-63):
    // lane = (channel-quad r, neighbor-group half of 8), b64 loads
    const int c0 = r * 4;
    float m[4] = {-1e30f, -1e30f, -1e30f, -1e30f};
    for (int k = half * 8; k < half * 8 + 8; ++k) {
        long q = (long)b * NPTS + sidx[wave][k];
        unsigned long long v =
            *reinterpret_cast<const unsigned long long*>(f_aggp + q * 64 + c0);
#pragma unroll
        for (int j = 0; j < 4; ++j)
            m[j] = fmaxf(m[j], bf2f((unsigned short)(v >> (16 * j))));
    }
#pragma unroll
    for (int j = 0; j < 4; ++j) m[j] = fmaxf(m[j], swap_half(m[j]));
    if (lane < 16) {
        unsigned long long o = 0;
#pragma unroll
        for (int j = 0; j < 4; ++j) o |= (unsigned long long)f2bf(m[j]) << (16 * j);
        *reinterpret_cast<unsigned long long*>(agg2 + p * 128 + c0) = o;
    }
}

// ---------------------------------------------------------------------------
// Final: out = leaky_relu(A1@W1 + b1 + A2@W2 + b2, 0.2), f32 out [M x 256]
// ---------------------------------------------------------------------------
__global__ __launch_bounds__(256) void gemm_dual_leaky_kernel(
        const unsigned short* __restrict__ A1, const unsigned short* __restrict__ W1t,
        const float* __restrict__ b1,
        const unsigned short* __restrict__ A2, const unsigned short* __restrict__ W2t,
        const float* __restrict__ b2,
        float* __restrict__ out) {
    constexpr int K = 128, NF = 256;
    __shared__ __align__(16) unsigned short sW1[64 * WPAD];
    __shared__ __align__(16) unsigned short sW2[64 * WPAD];

    const int tid  = threadIdx.x;
    const int wave = tid >> 5, lane = tid & 31;
    const int half = lane >> 4, r = lane & 15;
    const long row0 = ((long)blockIdx.x * 8 + wave) * 16;

    Frag16 aF1[4], aF2[4];
    const uint4* pA1 = reinterpret_cast<const uint4*>(A1 + (size_t)(row0 + r) * K);
    const uint4* pA2 = reinterpret_cast<const uint4*>(A2 + (size_t)(row0 + r) * K);
#pragma unroll
    for (int kt = 0; kt < 4; ++kt) {
        aF1[kt].q[0] = pA1[kt * 4 + half * 2]; aF1[kt].q[1] = pA1[kt * 4 + half * 2 + 1];
        aF2[kt].q[0] = pA2[kt * 4 + half * 2]; aF2[kt].q[1] = pA2[kt * 4 + half * 2 + 1];
    }

    for (int ng = 0; ng < NF / 64; ++ng) {
        __syncthreads();
        for (int ch = tid; ch < 64 * (K / 8); ch += 256) {
            int c = ch >> 4, cq = ch & 15;
            *reinterpret_cast<uint4*>(sW1 + c * WPAD + cq * 8) =
                reinterpret_cast<const uint4*>(W1t + (size_t)(ng * 64 + c) * K)[cq];
            *reinterpret_cast<uint4*>(sW2 + c * WPAD + cq * 8) =
                reinterpret_cast<const uint4*>(W2t + (size_t)(ng * 64 + c) * K)[cq];
        }
        __syncthreads();
#pragma unroll
        for (int t = 0; t < 4; ++t) {
            const int n0 = t * 16;
            v8f acc = {0.f, 0.f, 0.f, 0.f, 0.f, 0.f, 0.f, 0.f};
#pragma unroll
            for (int kt = 0; kt < 4; ++kt) {
                Frag16 wF;
                const unsigned short* base = sW1 + (n0 + r) * WPAD + kt * 32 + half * 16;
                wF.q[0] = *reinterpret_cast<const uint4*>(base);
                wF.q[1] = *reinterpret_cast<const uint4*>(base + 8);
                acc = wmma_bf16(aF1[kt], wF, acc);
            }
#pragma unroll
            for (int kt = 0; kt < 4; ++kt) {
                Frag16 wF;
                const unsigned short* base = sW2 + (n0 + r) * WPAD + kt * 32 + half * 16;
                wF.q[0] = *reinterpret_cast<const uint4*>(base);
                wF.q[1] = *reinterpret_cast<const uint4*>(base + 8);
                acc = wmma_bf16(aF2[kt], wF, acc);
            }
            const int col = ng * 64 + n0 + r;
            const float bv = b1[col] + b2[col];
#pragma unroll
            for (int i = 0; i < 8; ++i) {
                float v = acc[i] + bv;
                v = (v >= 0.f) ? v : 0.2f * v;   // leaky_relu(0.2)
                out[(size_t)(row0 + half * 8 + i) * NF + col] = v;
            }
        }
    }
}

// ---------------------------------------------------------------------------
// Host launcher
// ---------------------------------------------------------------------------
extern "C" void kernel_launch(void* const* d_in, const int* in_sizes, int n_in,
                              void* d_out, int out_size, void* d_ws, size_t ws_size,
                              hipStream_t stream) {
    (void)in_sizes; (void)n_in; (void)out_size; (void)ws_size;

    const float* feature = (const float*)d_in[0];
    const float* xyz     = (const float*)d_in[1];
    const int*   nidx    = (const int*)d_in[2];

    // params, setup_inputs() dict insertion order: each layer {w,gamma,beta,mean,var}
    // layers: mlp1, lfa_mlp1, pool1, lfa_mlp2, pool2, mlp2, shortcut
    const float* pw[7], *pg[7], *pb[7], *pm[7], *pv[7];
    for (int l = 0; l < 7; ++l) {
        pw[l] = (const float*)d_in[3 + l * 5 + 0];
        pg[l] = (const float*)d_in[3 + l * 5 + 1];
        pb[l] = (const float*)d_in[3 + l * 5 + 2];
        pm[l] = (const float*)d_in[3 + l * 5 + 3];
        pv[l] = (const float*)d_in[3 + l * 5 + 4];
    }
    const int cin[7]  = {128, 13, 128, 64, 128, 128, 128};
    const int cout[7] = { 64, 64,  64, 64, 128, 256, 256};

    // workspace carve-out (256B aligned)
    char* ws = (char*)d_ws;
    size_t off = 0;
    auto carve = [&](size_t bytes) -> char* {
        char* p = ws + off;
        off = (off + bytes + 255) & ~(size_t)255;
        return p;
    };
    unsigned short* featbf = (unsigned short*)carve((size_t)MPTS * 128 * 2);
    unsigned short* f_pc   = (unsigned short*)carve((size_t)MPTS * 64 * 2);
    unsigned short* agg1   = (unsigned short*)carve((size_t)MPTS * 128 * 2);
    unsigned short* f_aggp = (unsigned short*)carve((size_t)MPTS * 64 * 2);
    unsigned short* agg2   = (unsigned short*)carve((size_t)MPTS * 128 * 2);
    unsigned short* f_lfa  = (unsigned short*)carve((size_t)MPTS * 128 * 2);
    unsigned short* wbf[7], *wbf_t[7];
    float* bias[7];
    for (int l = 0; l < 7; ++l) {
        wbf[l]   = (unsigned short*)carve((size_t)cin[l] * cout[l] * 2);
        wbf_t[l] = (unsigned short*)carve((size_t)cin[l] * cout[l] * 2);
        bias[l]  = (float*)carve((size_t)cout[l] * 4);
    }

    // 1) fold BN into weights/bias (row-major + transposed bf16)
    for (int l = 0; l < 7; ++l) {
        int total = cin[l] * cout[l];
        fold_bn_kernel<<<(total + 255) / 256, 256, 0, stream>>>(
            pw[l], pg[l], pb[l], pm[l], pv[l], wbf[l], wbf_t[l], bias[l],
            total, cin[l], cout[l]);
    }
    // 2) feature -> bf16 (paired)
    {
        int npairs = MPTS * 128 / 2;
        cvt_bf16_kernel<<<(npairs + 255) / 256, 256, 0, stream>>>(
            feature, (unsigned*)featbf, npairs);
    }
    const int gemm_blocks = MPTS / 128;   // 640; 8 waves x 16 rows each

    // 3) mlp1: f_pc = relu(bn(feature @ W))           [M x 64]
    gemm_bf16_kernel<64, true><<<gemm_blocks, 256, 0, stream>>>(featbf, wbf_t[0], bias[0], f_pc);
    // 4) rel-pos + lfa_mlp1 + gather/max -> agg1      [M x 128]
    lfa1_kernel<<<MPTS / 4, 256, 0, stream>>>(xyz, nidx, wbf[1], bias[1], f_pc, agg1);
    // 5) pool1: f_aggp = relu(bn(agg1 @ W))           [M x 64]
    gemm_bf16_kernel<64, true><<<gemm_blocks, 256, 0, stream>>>(agg1, wbf_t[2], bias[2], f_aggp);
    // 6) fused lfa_mlp2 (WMMA) + gather/max -> agg2   [M x 128]
    lfa2_kernel<<<MPTS / 8, 256, 0, stream>>>(xyz, nidx, wbf[1], bias[1],
                                              wbf_t[3], bias[3], f_aggp, agg2);
    // 7) pool2: f_lfa = relu(bn(agg2 @ W))            [M x 128]
    gemm_bf16_kernel<128, true><<<gemm_blocks, 256, 0, stream>>>(agg2, wbf_t[4], bias[4], f_lfa);
    // 8) mlp2 + shortcut + leaky_relu -> out (f32)    [M x 256]
    gemm_dual_leaky_kernel<<<gemm_blocks, 256, 0, stream>>>(
        f_lfa, wbf_t[5], bias[5], featbf, wbf_t[6], bias[6], (float*)d_out);
}
